// TransLayer_8796093022313
// MI455X (gfx1250) — matrix operand order repaired and checked by reading, not compiled, not run on hardware
//
#include <hip/hip_runtime.h>
#include <float.h>

typedef __bf16 bf16_t;
typedef __attribute__((ext_vector_type(16))) __bf16 v16bf;
typedef __attribute__((ext_vector_type(8)))  __bf16 bf16x8;
typedef __attribute__((ext_vector_type(8)))  float  v8f;
typedef __attribute__((ext_vector_type(2)))  float  v2f;

// Problem constants
// B=4, N=8192, DIM=512, HEADS=8, DH=64, M(landmarks)=256, pad=0, l=32, BH=32

__device__ inline float bf2f(bf16_t x) {
  unsigned short u = __builtin_bit_cast(unsigned short, x);
  unsigned int v = ((unsigned int)u) << 16;
  return __builtin_bit_cast(float, v);
}
__device__ inline bf16_t f2bf(float f) {
  unsigned int u = __builtin_bit_cast(unsigned int, f);
  u += 0x7fffu + ((u >> 16) & 1u);          // round-to-nearest-even
  unsigned short s = (unsigned short)(u >> 16);
  return __builtin_bit_cast(bf16_t, s);
}
__device__ inline float wave_max(float v){
  #pragma unroll
  for (int m = 16; m > 0; m >>= 1) v = fmaxf(v, __shfl_xor(v, m, 32));
  return v;
}
__device__ inline float wave_sum(float v){
  #pragma unroll
  for (int m = 16; m > 0; m >>= 1) v += __shfl_xor(v, m, 32);
  return v;
}

// gfx1250 async global->LDS copy (ASYNCcnt-tracked, no VGPR round trip).
__device__ inline void async_copy_b128(void* lds_dst, const void* gsrc){
#if defined(__gfx1250__)
  unsigned int       lds = (unsigned int)(unsigned long long)lds_dst;
  unsigned long long g   = (unsigned long long)gsrc;
  asm volatile("global_load_async_to_lds_b128 %0, %1, off"
               :: "v"(lds), "v"(g) : "memory");
#else
  *(uint4*)lds_dst = *(const uint4*)gsrc;
#endif
}
// Async loads complete in order -> waiting "<= N" retires everything older
// than the newest N outstanding ops (software pipelining). N is compile-time.
template<int N>
__device__ inline void async_wait_lt(){
#if defined(__gfx1250__)
  if constexpr (N == 4)      asm volatile("s_wait_asynccnt 0x4" ::: "memory");
  else if constexpr (N == 3) asm volatile("s_wait_asynccnt 0x3" ::: "memory");
  else if constexpr (N == 2) asm volatile("s_wait_asynccnt 0x2" ::: "memory");
  else                       asm volatile("s_wait_asynccnt 0x0" ::: "memory");
#endif
}

// ---------------- small elementwise kernels ----------------

// w_qkv [512,1536] -> transposed bf16 [1536,512], q columns pre-scaled by 1/8
__global__ void k_convert_wqkv_T(const float* __restrict__ w, bf16_t* __restrict__ o){
  int i = blockIdx.x * 256 + threadIdx.x;    // i = c*512 + k
  if (i >= 1536 * 512) return;
  int c = i >> 9, k = i & 511;
  float v = w[k * 1536 + c];
  if (c < 512) v *= 0.125f;
  o[i] = f2bf(v);
}

// w_out [512,512] -> transposed bf16 [512,512]
__global__ void k_convert_wout_T(const float* __restrict__ w, bf16_t* __restrict__ o){
  int i = blockIdx.x * 256 + threadIdx.x;    // i = n*512 + k
  if (i >= 512 * 512) return;
  int n = i >> 9, k = i & 511;
  o[i] = f2bf(w[k * 512 + n]);
}

// vh [bh,8192,64] -> vt [bh,64,8192] (so attn3@v can use the async transB path)
__global__ void k_transpose_vh(const bf16_t* __restrict__ v, bf16_t* __restrict__ vt){
  long long i = (long long)blockIdx.x * 256 + threadIdx.x;   // 16,777,216
  long long bh = i >> 19; int rem = (int)(i & 524287);
  int d = rem >> 13, n = rem & 8191;
  vt[i] = v[bh * 524288 + (long long)n * 64 + d];
}

// t2f [bh,256,64] f32 -> t2bT [bh,64,256] bf16
__global__ void k_transpose_t2(const float* __restrict__ t2f, bf16_t* __restrict__ o){
  long long i = (long long)blockIdx.x * 256 + threadIdx.x;   // 524,288
  long long bh = i >> 14; int rem = (int)(i & 16383);
  int d = rem >> 8, m = rem & 255;
  o[i] = f2bf(t2f[bh * 16384 + (long long)m * 64 + d]);
}

__global__ __launch_bounds__(256) void k_layernorm(const float* __restrict__ x,
                                                   const float* __restrict__ w,
                                                   const float* __restrict__ b,
                                                   bf16_t* __restrict__ o){
  __shared__ float red[256];
  long long row = blockIdx.x;            // 32768 rows of 512
  const float* p = x + row * 512;
  int t = threadIdx.x;
  float a0 = p[t], a1 = p[t + 256];
  red[t] = a0 + a1; __syncthreads();
  for (int s = 128; s > 0; s >>= 1){ if (t < s) red[t] += red[t + s]; __syncthreads(); }
  float mu = red[0] * (1.f / 512.f);
  __syncthreads();
  float d0 = a0 - mu, d1 = a1 - mu;
  red[t] = d0 * d0 + d1 * d1; __syncthreads();
  for (int s = 128; s > 0; s >>= 1){ if (t < s) red[t] += red[t + s]; __syncthreads(); }
  float inv = rsqrtf(red[0] * (1.f / 512.f) + 1e-5f);
  o[row * 512 + t]       = f2bf(d0 * inv * w[t]       + b[t]);
  o[row * 512 + t + 256] = f2bf(d1 * inv * w[t + 256] + b[t + 256]);
}

__global__ void k_landmarks(const bf16_t* __restrict__ src, float* __restrict__ of,
                            bf16_t* __restrict__ ob){
  // grid: BH*256 blocks, 64 threads; landmark m averages tokens m*32..m*32+31
  int blk = blockIdx.x; int bh = blk >> 8; int m = blk & 255; int d = threadIdx.x;
  const bf16_t* p = src + (((long long)bh * 8192) + (long long)m * 32) * 64 + d;
  float s = 0.f;
  #pragma unroll
  for (int j = 0; j < 32; ++j) s += bf2f(p[(long long)j * 64]);
  s *= (1.f / 32.f);
  long long o = (((long long)bh * 256) + m) * 64 + d;
  of[o] = s; ob[o] = f2bf(s);
}

__global__ void k_softmax256(const float* __restrict__ S, float* __restrict__ Of,
                             bf16_t* __restrict__ Ob){
  long long row = blockIdx.x;            // one wave per 256-wide row
  const float* p = S + row * 256;
  int lane = threadIdx.x;
  float v[8]; float mx = -FLT_MAX;
  #pragma unroll
  for (int i = 0; i < 8; ++i){ v[i] = p[lane + i * 32]; mx = fmaxf(mx, v[i]); }
  mx = wave_max(mx);
  float s = 0.f;
  #pragma unroll
  for (int i = 0; i < 8; ++i){ v[i] = __expf(v[i] - mx); s += v[i]; }
  s = wave_sum(s);
  float inv = 1.f / s;
  #pragma unroll
  for (int i = 0; i < 8; ++i){
    long long idx = row * 256 + lane + i * 32;
    float o = v[i] * inv;
    if (Of) Of[idx] = o;
    if (Ob) Ob[idx] = f2bf(o);
  }
}

__global__ __launch_bounds__(256) void k_softmax_big(const float* __restrict__ S,
                                                     bf16_t* __restrict__ O, int len){
  __shared__ float red[256];
  long long row = blockIdx.x;            // len = 8192
  const float* p = S + row * (long long)len;
  int t = threadIdx.x;
  float v[32]; float mx = -FLT_MAX;
  #pragma unroll
  for (int i = 0; i < 32; ++i){ v[i] = p[t + i * 256]; mx = fmaxf(mx, v[i]); }
  red[t] = mx; __syncthreads();
  for (int s = 128; s > 0; s >>= 1){ if (t < s) red[t] = fmaxf(red[t], red[t + s]); __syncthreads(); }
  mx = red[0]; __syncthreads();
  float sum = 0.f;
  #pragma unroll
  for (int i = 0; i < 32; ++i){ v[i] = __expf(v[i] - mx); sum += v[i]; }
  red[t] = sum; __syncthreads();
  for (int s = 128; s > 0; s >>= 1){ if (t < s) red[t] += red[t + s]; __syncthreads(); }
  float inv = 1.f / red[0];
  #pragma unroll
  for (int i = 0; i < 32; ++i) O[row * (long long)len + t + i * 256] = f2bf(v[i] * inv);
}

__global__ void k_sumabs(const float* __restrict__ A, float* __restrict__ out,
                         int idx_stride, int k_stride){
  int blk = blockIdx.x; int bh = blk >> 8; int idx = blk & 255; int lane = threadIdx.x;
  const float* p = A + ((long long)bh << 16) + (long long)idx * idx_stride;
  float s = 0.f;
  for (int j = lane; j < 256; j += 32) s += fabsf(p[(long long)j * k_stride]);
  s = wave_sum(s);
  if (lane == 0) out[blk] = s;
}

__global__ __launch_bounds__(256) void k_maxreduce(const float* __restrict__ in, int n,
                                                   float* __restrict__ out){
  __shared__ float red[256];
  float m = -FLT_MAX;
  for (int i = threadIdx.x; i < n; i += 256) m = fmaxf(m, in[i]);
  red[threadIdx.x] = m; __syncthreads();
  for (int s = 128; s > 0; s >>= 1){
    if (threadIdx.x < s) red[threadIdx.x] = fmaxf(red[threadIdx.x], red[threadIdx.x + s]);
    __syncthreads();
  }
  if (threadIdx.x == 0) *out = red[0];
}

__global__ void k_zinit(const float* __restrict__ A2, float* __restrict__ Z,
                        const float* __restrict__ mx2){
  long long idx = (long long)blockIdx.x * 256 + threadIdx.x;   // 32*256*256 elems
  float inv = 1.f / (mx2[0] * mx2[1]);
  long long bh = idx >> 16; int r = (int)((idx >> 8) & 255); int c = (int)(idx & 255);
  Z[idx] = A2[(bh << 16) + ((long long)c << 8) + r] * inv;     // transpose / scale
}

__global__ void k_aI_minus(float* __restrict__ D, const float* __restrict__ S, float a){
  long long idx = (long long)blockIdx.x * 256 + threadIdx.x;
  int r = (int)((idx >> 8) & 255), c = (int)(idx & 255);
  D[idx] = ((r == c) ? a : 0.f) - S[idx];
}

// ---------------- generic bf16 WMMA GEMM (double-buffered async pipeline) ----------------

struct EpiParams {
  int mode;                 // 0:f32 C, 1:bf16 C, 2:qkv scatter, 3:final+bias+x, 4:attn-out+conv
  float* cf; bf16_t* cb;
  long long ldc, cbs;
  bf16_t *qh, *kh, *vh;           // mode 2
  const float* xres; const float* bias;   // mode 3
  const bf16_t* vconv; const float* resw; // mode 4
};

#define GBM 128
#define GBN 64
#define GBK 32

template<int TRANSB>
__global__ __launch_bounds__(256) void k_gemm_bf16(
    const bf16_t* __restrict__ A, long long lda, long long a_bs,
    const bf16_t* __restrict__ B, long long ldb, long long b_bs,
    int Mdim, int Ndim, int Kdim, EpiParams ep)
{
  __shared__ __align__(16) bf16_t As[2][GBM * GBK];   // 2 x 8 KB, row-major [m][k]
  __shared__ __align__(16) bf16_t Bs[2][GBN * GBK];   // 2 x 4 KB, transposed [n][k]
  constexpr int BATCH = TRANSB ? 3 : 2;               // async ops per stage per wave
  int bz = blockIdx.z;
  const bf16_t* Ab = A + (long long)bz * a_bs;
  const bf16_t* Bb = B + (long long)bz * b_bs;
  int m0 = blockIdx.y * GBM, n0 = blockIdx.x * GBN;
  int tid = threadIdx.x, w = tid >> 5, lane = tid & 31;
  int wm = (w & 3) * 32, wn = (w >> 2) * 32;       // 4x2 wave grid, 32x32 tile per wave
  int hi = lane >> 4, lr = lane & 15;
  v8f acc[2][2];
  #pragma unroll
  for (int mi = 0; mi < 2; ++mi)
    #pragma unroll
    for (int ni = 0; ni < 2; ++ni)
      #pragma unroll
      for (int r = 0; r < 8; ++r) acc[mi][ni][r] = 0.f;

  auto stage = [&](int buf, int kk0){
    // A tile: 128x32 bf16 = 512 x 16B, exactly 2 async copies per thread
    #pragma unroll
    for (int ii = 0; ii < 2; ++ii){
      int i = tid + ii * 256;
      int r = i >> 2, cs = (i & 3) * 8;
      async_copy_b128(&As[buf][r * GBK + cs],
                      Ab + (long long)(m0 + r) * lda + kk0 + cs);
    }
    if constexpr (TRANSB){                         // B is [N,K]: 1 async copy per thread
      int n = tid >> 2, cs = (tid & 3) * 8;
      async_copy_b128(&Bs[buf][n * GBK + cs],
                      Bb + (long long)(n0 + n) * ldb + kk0 + cs);
    } else {                                       // B is [K,N]: vector load + register scatter
      int k = tid >> 3, n = (tid & 7) * 8;
      bf16x8 tmp = *(const bf16x8*)(Bb + (long long)(kk0 + k) * ldb + n0 + n);
      #pragma unroll
      for (int j = 0; j < 8; ++j) Bs[buf][(n + j) * GBK + k] = tmp[j];
    }
    if (kk0 + GBK < Kdim)                          // prefetch tile after the staged one (WGP scope)
      __builtin_prefetch(Ab + (long long)(m0 + (tid >> 1)) * lda + kk0 + GBK, 0, 3);
  };

  int nk = Kdim / GBK;
  stage(0, 0);
  for (int t = 0; t < nk; ++t){
    int cur = t & 1;
    bool more = (t + 1) < nk;
    if (more){
      stage(1 - cur, (t + 1) * GBK);               // overlaps with compute of tile t
      async_wait_lt<BATCH>();                      // retire tile t's async batch
    } else {
      async_wait_lt<0>();
    }
    __syncthreads();

    const bf16_t* Ac = As[cur];
    const bf16_t* Bc = Bs[cur];
    v16bf af[2], bfv[2];
    #pragma unroll
    for (int mi = 0; mi < 2; ++mi){
      // 16-bit A 16x32 layout: low lanes hold K 0-7 & 16-23, high lanes +8
      const bf16_t* ap = &Ac[(wm + mi * 16 + lr) * GBK + hi * 8];
      #pragma unroll
      for (int i = 0; i < 8; ++i){ af[mi][i] = ap[i]; af[mi][8 + i] = ap[16 + i]; }
    }
    #pragma unroll
    for (int ni = 0; ni < 2; ++ni){
      // 16-bit B 32x16 layout: lane = column, low lanes K 0-7 & 16-23, high lanes +8
      const bf16_t* bp = &Bc[(wn + ni * 16 + lr) * GBK + hi * 8];
      #pragma unroll
      for (int i = 0; i < 8; ++i){ bfv[ni][i] = bp[i]; bfv[ni][8 + i] = bp[16 + i]; }
    }
    #pragma unroll
    for (int mi = 0; mi < 2; ++mi)
      #pragma unroll
      for (int ni = 0; ni < 2; ++ni)
        acc[mi][ni] = __builtin_amdgcn_wmma_f32_16x16x32_bf16(
            false, af[mi], false, bfv[ni], (short)0, acc[mi][ni], false, false);
    __syncthreads();                               // all waves done reading buffer `cur`
  }

  #pragma unroll
  for (int mi = 0; mi < 2; ++mi)
  #pragma unroll
  for (int ni = 0; ni < 2; ++ni)
  #pragma unroll
  for (int r = 0; r < 8; ++r){
    int m = m0 + wm + mi * 16 + r + 8 * hi;     // C layout: lanes16-31 -> M+8
    int n = n0 + wn + ni * 16 + lr;
    float val = acc[mi][ni][r];
    if (ep.mode == 0){
      ep.cf[(long long)bz * ep.cbs + (long long)m * ep.ldc + n] = val;
    } else if (ep.mode == 1){
      ep.cb[(long long)bz * ep.cbs + (long long)m * ep.ldc + n] = f2bf(val);
    } else if (ep.mode == 2){                    // qkv -> per-head [b*8+h, n, 64]
      int which = n >> 9, h = (n >> 6) & 7, dh = n & 63;
      int bb = m >> 13, tok = m & 8191;
      bf16_t* dst = (which == 0) ? ep.qh : ((which == 1) ? ep.kh : ep.vh);
      dst[(((long long)(bb * 8 + h)) * 8192 + tok) * 64 + dh] = f2bf(val);
    } else if (ep.mode == 3){                    // y = acc + b_out + x
      long long o = (long long)m * 512 + n;
      ep.cf[o] = val + ep.bias[n] + ep.xres[o];
    } else {                                     // mode 4: + depthwise conv(v), scatter to [b,n,512]
      int h = bz & 7;
      const bf16_t* vp = ep.vconv + (long long)bz * 8192 * 64 + n;
      float rr = 0.f;
      #pragma unroll
      for (int j = 0; j < 33; ++j){
        int t = m + j - 16;
        if (t >= 0 && t < 8192) rr += ep.resw[h * 33 + j] * bf2f(vp[(long long)t * 64]);
      }
      ep.cb[(((long long)(bz >> 3)) * 8192 + m) * 512 + h * 64 + n] = f2bf(val + rr);
    }
  }
}

// ---------------- fp32 WMMA GEMM (pinv chain): D = c0*I + c1*(A@B) ----------------

#define FBM 64
#define FBN 64
#define FBK 16

template<int TRANSB>
__global__ __launch_bounds__(128) void k_gemm_f32(
    const float* __restrict__ A, long long lda, long long a_bs,
    const float* __restrict__ B, long long ldb, long long b_bs,
    float* __restrict__ D, long long ldd, long long d_bs,
    int Mdim, int Ndim, int Kdim, float c0, float c1)
{
  __shared__ __align__(16) float As[2][FBM * FBK];   // 2 x 4 KB [m][k]
  __shared__ __align__(16) float Bs[2][FBN * FBK];   // 2 x 4 KB [n][k]
  constexpr int BATCH = TRANSB ? 4 : 2;
  int bz = blockIdx.z;
  const float* Ab = A + (long long)bz * a_bs;
  const float* Bb = B + (long long)bz * b_bs;
  int m0 = blockIdx.y * FBM, n0 = blockIdx.x * FBN;
  int tid = threadIdx.x, w = tid >> 5, lane = tid & 31;
  int wm = (w & 1) * 32, wn = (w >> 1) * 32;      // 2x2 wave grid
  int hi = lane >> 4, lr = lane & 15;
  v8f acc[2][2];
  #pragma unroll
  for (int mi = 0; mi < 2; ++mi)
    #pragma unroll
    for (int ni = 0; ni < 2; ++ni)
      #pragma unroll
      for (int r = 0; r < 8; ++r) acc[mi][ni][r] = 0.f;

  auto stage = [&](int buf, int kk0){
    // A tile: 64x16 f32 = 256 x 16B, 2 async copies per thread
    #pragma unroll
    for (int ii = 0; ii < 2; ++ii){
      int i = tid + ii * 128;
      int r = i >> 2, cs = (i & 3) * 4;
      async_copy_b128(&As[buf][r * FBK + cs],
                      Ab + (long long)(m0 + r) * lda + kk0 + cs);
    }
    if constexpr (TRANSB){
      #pragma unroll
      for (int ii = 0; ii < 2; ++ii){
        int i = tid + ii * 128;
        int n = i >> 2, cs = (i & 3) * 4;
        async_copy_b128(&Bs[buf][n * FBK + cs],
                        Bb + (long long)(n0 + n) * ldb + kk0 + cs);
      }
    } else {                                       // vector load + register scatter
      #pragma unroll
      for (int ii = 0; ii < 2; ++ii){
        int i = tid + ii * 128;
        int k = i >> 4, n = (i & 15) * 4;
        float4 tmp = *(const float4*)(Bb + (long long)(kk0 + k) * ldb + n0 + n);
        Bs[buf][(n + 0) * FBK + k] = tmp.x;
        Bs[buf][(n + 1) * FBK + k] = tmp.y;
        Bs[buf][(n + 2) * FBK + k] = tmp.z;
        Bs[buf][(n + 3) * FBK + k] = tmp.w;
      }
    }
    if (kk0 + FBK < Kdim)
      __builtin_prefetch(Ab + (long long)(m0 + (tid >> 1)) * lda + kk0 + FBK, 0, 3);
  };

  int nk = Kdim / FBK;
  stage(0, 0);
  for (int t = 0; t < nk; ++t){
    int cur = t & 1;
    bool more = (t + 1) < nk;
    if (more){
      stage(1 - cur, (t + 1) * FBK);
      async_wait_lt<BATCH>();
    } else {
      async_wait_lt<0>();
    }
    __syncthreads();
    const float* Ac = As[cur];
    const float* Bc = Bs[cur];
#if __has_builtin(__builtin_amdgcn_wmma_f32_16x16x4_f32)
    #pragma unroll
    for (int ks = 0; ks < FBK; ks += 4){
      v2f a[2], bv[2];
      #pragma unroll
      for (int mi = 0; mi < 2; ++mi){
        // 32-bit A 16x4: low lanes K={0,1}, high lanes K={2,3}
        const float* ap = &Ac[(wm + mi * 16 + lr) * FBK + ks + hi * 2];
        a[mi][0] = ap[0]; a[mi][1] = ap[1];
      }
      #pragma unroll
      for (int ni = 0; ni < 2; ++ni){
        const float* bp = &Bc[(wn + ni * 16 + lr) * FBK + ks + hi * 2];
        bv[ni][0] = bp[0]; bv[ni][1] = bp[1];
      }
      #pragma unroll
      for (int mi = 0; mi < 2; ++mi)
        #pragma unroll
        for (int ni = 0; ni < 2; ++ni)
          acc[mi][ni] = __builtin_amdgcn_wmma_f32_16x16x4_f32(
              false, a[mi], false, bv[ni], (short)0, acc[mi][ni], false, false);
    }
#else
    for (int kk = 0; kk < FBK; ++kk)
      #pragma unroll
      for (int mi = 0; mi < 2; ++mi)
        #pragma unroll
        for (int ni = 0; ni < 2; ++ni){
          float bvv = Bc[(wn + ni * 16 + lr) * FBK + kk];
          #pragma unroll
          for (int r = 0; r < 8; ++r)
            acc[mi][ni][r] += Ac[(wm + mi * 16 + r + 8 * hi) * FBK + kk] * bvv;
        }
#endif
    __syncthreads();
  }

  #pragma unroll
  for (int mi = 0; mi < 2; ++mi)
  #pragma unroll
  for (int ni = 0; ni < 2; ++ni)
  #pragma unroll
  for (int r = 0; r < 8; ++r){
    int m = m0 + wm + mi * 16 + r + 8 * hi;
    int n = n0 + wn + ni * 16 + lr;
    float val = c1 * acc[mi][ni][r] + ((m == n) ? c0 : 0.f);
    D[(long long)bz * d_bs + (long long)m * ldd + n] = val;
  }
}

// ---------------- host orchestration ----------------

extern "C" void kernel_launch(void* const* d_in, const int* in_sizes, int n_in,
                              void* d_out, int out_size, void* d_ws, size_t ws_size,
                              hipStream_t stream)
{
  const float* x     = (const float*)d_in[0];
  const float* ln_w  = (const float*)d_in[1];
  const float* ln_b  = (const float*)d_in[2];
  const float* w_qkv = (const float*)d_in[3];
  const float* w_out = (const float*)d_in[4];
  const float* b_out = (const float*)d_in[5];
  const float* res_w = (const float*)d_in[6];
  float* out = (float*)d_out;
  (void)in_sizes; (void)n_in; (void)out_size; (void)ws_size;

  char* base = (char*)d_ws;
  size_t off = 0;
  auto walloc = [&](size_t bytes) -> char* {
    char* p = base + off;
    off = (off + bytes + 255) & ~(size_t)255;
    return p;
  };

  bf16_t* xn    = (bf16_t*)walloc(16777216ull * 2);  // LN(x) bf16 [B*N,512]
  bf16_t* wqbT  = (bf16_t*)walloc(786432ull   * 2);  // w_qkv^T bf16 [1536,512]
  bf16_t* wobT  = (bf16_t*)walloc(262144ull   * 2);  // w_out^T bf16 [512,512]
  bf16_t* qh    = (bf16_t*)walloc(16777216ull * 2);  // [bh,8192,64]
  bf16_t* kh    = (bf16_t*)walloc(16777216ull * 2);
  bf16_t* vh    = (bf16_t*)walloc(16777216ull * 2);
  bf16_t* vt    = (bf16_t*)walloc(16777216ull * 2);  // v^T [bh,64,8192]
  float*  qlf   = (float*) walloc(524288ull * 4);    // landmarks fp32 [bh,256,64]
  float*  klf   = (float*) walloc(524288ull * 4);
  bf16_t* qlb   = (bf16_t*)walloc(524288ull * 2);
  bf16_t* klb   = (bf16_t*)walloc(524288ull * 2);
  float*  simf  = (float*) walloc(67108864ull * 4);  // sim3 then sim1 (fp32)
  bf16_t* attnb = (bf16_t*)walloc(67108864ull * 2);  // attn3 then attn1 (bf16)
  float*  A2    = (float*) walloc(2097152ull * 4);   // attn2 [bh,256,256]
  float*  Z0    = (float*) walloc(2097152ull * 4);
  float*  Z1    = (float*) walloc(2097152ull * 4);
  float*  XZ    = (float*) walloc(2097152ull * 4);
  float*  Pb    = (float*) walloc(2097152ull * 4);
  float*  Rb    = (float*) walloc(2097152ull * 4);
  float*  t1    = (float*) walloc(524288ull * 4);    // attn3@v
  float*  t2f   = (float*) walloc(524288ull * 4);    // pinv @ t1
  bf16_t* t2bT  = (bf16_t*)walloc(524288ull * 2);    // t2^T bf16 [bh,64,256]
  bf16_t* oin   = (bf16_t*)walloc(16777216ull * 2);  // [b,n,512] pre-projection
  float*  csum  = (float*) walloc(8192ull * 4);
  float*  rsum  = (float*) walloc(8192ull * 4);
  float*  mx2   = (float*) walloc(256);

  // 0) weight conversions (transposed) + LN
  k_convert_wqkv_T<<<3072, 256, 0, stream>>>(w_qkv, wqbT);
  k_convert_wout_T<<<1024, 256, 0, stream>>>(w_out, wobT);
  k_layernorm<<<32768, 256, 0, stream>>>(x, ln_w, ln_b, xn);

  // 1) qkv = xn @ w_qkv  (B^T, async path; scatter into per-head layouts)
  EpiParams ep{};
  ep.mode = 2; ep.qh = qh; ep.kh = kh; ep.vh = vh;
  k_gemm_bf16<1><<<dim3(24, 256, 1), 256, 0, stream>>>(xn, 512, 0, wqbT, 512, 0,
                                                       32768, 1536, 512, ep);
  k_transpose_vh<<<65536, 256, 0, stream>>>(vh, vt);

  // 2) landmark means
  k_landmarks<<<8192, 64, 0, stream>>>(qh, qlf, qlb);
  k_landmarks<<<8192, 64, 0, stream>>>(kh, klf, klb);

  // 3) attn2 = softmax(q_l @ k_l^T)   (fp32 WMMA)
  k_gemm_f32<1><<<dim3(4, 4, 32), 128, 0, stream>>>(qlf, 64, 16384, klf, 64, 16384,
                                                    A2, 256, 65536,
                                                    256, 256, 64, 0.f, 1.f);
  k_softmax256<<<8192, 32, 0, stream>>>(A2, A2, nullptr);

  // 4) pinv init: Z = A2^T / (max colsum * max rowsum)
  k_sumabs<<<8192, 32, 0, stream>>>(A2, csum, 256, 1);
  k_sumabs<<<8192, 32, 0, stream>>>(A2, rsum, 1, 256);
  k_maxreduce<<<1, 256, 0, stream>>>(csum, 8192, mx2 + 0);
  k_maxreduce<<<1, 256, 0, stream>>>(rsum, 8192, mx2 + 1);
  k_zinit<<<8192, 256, 0, stream>>>(A2, Z0, mx2);

  // 5) Newton-Schulz: Z = 0.25*Z@(13I - XZ@(15I - XZ@(7I - XZ)))
  float* Zc = Z0; float* Za = Z1;
  for (int it = 0; it < 6; ++it){
    k_gemm_f32<0><<<dim3(4, 4, 32), 128, 0, stream>>>(A2, 256, 65536, Zc, 256, 65536,
                                                      XZ, 256, 65536,
                                                      256, 256, 256, 0.f, 1.f);
    k_aI_minus<<<8192, 256, 0, stream>>>(Pb, XZ, 7.f);                       // P = 7I - XZ
    k_gemm_f32<0><<<dim3(4, 4, 32), 128, 0, stream>>>(XZ, 256, 65536, Pb, 256, 65536,
                                                      Rb, 256, 65536,
                                                      256, 256, 256, 15.f, -1.f); // R = 15I - XZ@P
    k_gemm_f32<0><<<dim3(4, 4, 32), 128, 0, stream>>>(XZ, 256, 65536, Rb, 256, 65536,
                                                      Pb, 256, 65536,
                                                      256, 256, 256, 13.f, -1.f); // T = 13I - XZ@R
    k_gemm_f32<0><<<dim3(4, 4, 32), 128, 0, stream>>>(Zc, 256, 65536, Pb, 256, 65536,
                                                      Za, 256, 65536,
                                                      256, 256, 256, 0.f, 0.25f); // Znew
    float* t = Zc; Zc = Za; Za = t;
  }

  // 6) attn3 = softmax(q_l @ k^T, axis=-1)  -> bf16
  ep = EpiParams{}; ep.mode = 0; ep.cf = simf; ep.ldc = 8192; ep.cbs = 2097152;
  k_gemm_bf16<1><<<dim3(128, 2, 32), 256, 0, stream>>>(qlb, 64, 16384, kh, 64, 524288,
                                                       256, 8192, 64, ep);
  k_softmax_big<<<8192, 256, 0, stream>>>(simf, attnb, 8192);

  // 7) t1 = attn3 @ v (B = v^T, async path) ; t2 = pinv @ t1 ; transpose t2
  ep = EpiParams{}; ep.mode = 0; ep.cf = t1; ep.ldc = 64; ep.cbs = 16384;
  k_gemm_bf16<1><<<dim3(1, 2, 32), 256, 0, stream>>>(attnb, 8192, 2097152, vt, 8192, 524288,
                                                     256, 64, 8192, ep);
  k_gemm_f32<0><<<dim3(1, 4, 32), 128, 0, stream>>>(Zc, 256, 65536, t1, 64, 16384,
                                                    t2f, 64, 16384,
                                                    256, 64, 256, 0.f, 1.f);
  k_transpose_t2<<<2048, 256, 0, stream>>>(t2f, t2bT);

  // 8) attn1 = softmax(q @ k_l^T) -> bf16
  ep = EpiParams{}; ep.mode = 0; ep.cf = simf; ep.ldc = 256; ep.cbs = 2097152;
  k_gemm_bf16<1><<<dim3(4, 64, 32), 256, 0, stream>>>(qh, 64, 524288, klb, 64, 16384,
                                                      8192, 256, 64, ep);
  k_softmax256<<<262144, 32, 0, stream>>>(simf, nullptr, attnb);

  // 9) out_h = attn1 @ t2 + depthwise_conv(v)  -> oin [b,n,512]
  ep = EpiParams{}; ep.mode = 4; ep.cb = oin; ep.vconv = vh; ep.resw = res_w;
  k_gemm_bf16<1><<<dim3(1, 64, 32), 256, 0, stream>>>(attnb, 256, 2097152, t2bT, 256, 16384,
                                                      8192, 64, 256, ep);

  // 10) y = x + oin @ w_out + b_out  (B^T, async path)
  ep = EpiParams{}; ep.mode = 3; ep.cf = out; ep.xres = x; ep.bias = b_out;
  k_gemm_bf16<1><<<dim3(8, 256, 1), 256, 0, stream>>>(oin, 512, 0, wobT, 512, 0,
                                                      32768, 512, 512, ep);
}
